// cmSimpleRNNCell_21629455302684
// MI455X (gfx1250) — matrix operand order repaired and checked by reading, not compiled, not run on hardware
//
#include <hip/hip_runtime.h>
#include <hip/hip_bf16.h>
#include <stdint.h>

// ---------------- problem sizes ----------------
#define BATCH   64
#define D_IN    512
#define UNITS   1024
#define TOPICS  20
#define VOCAB   32000

typedef __attribute__((ext_vector_type(2)))  float        v2f;
typedef __attribute__((ext_vector_type(4)))  float        v4f;
typedef __attribute__((ext_vector_type(8)))  float        v8f;
typedef __attribute__((ext_vector_type(4)))  unsigned int v4u;
typedef __attribute__((ext_vector_type(8)))  unsigned int v8u;
typedef __attribute__((ext_vector_type(16))) __bf16       v16bf;

#define USE_TDM 1   // Tensor Data Mover for the A-matrix stage in k3

// =====================================================================
// Kernel 1: next_state = inputs@kernel + prev_state@recurrent_kernel
// Full f32 precision via V_WMMA_F32_16X16X4_F32 (wave32).
// =====================================================================
__global__ __launch_bounds__(128) void k1_next_state(
    const float* __restrict__ x,   // [64][512]
    const float* __restrict__ h,   // [64][1024]
    const float* __restrict__ Wx,  // [512][1024]
    const float* __restrict__ Wh,  // [1024][1024]
    float* __restrict__ ns)        // [64][1024]  (output tail)
{
    const int wave = threadIdx.x >> 5;
    const int lane = threadIdx.x & 31;
    const int lo = lane & 15, hi = lane >> 4;
    const int n = blockIdx.x * 16 + lo;   // output column
    const int m = wave * 16 + lo;         // A row for this lane

    v8f acc = {};
    for (int k = 0; k < D_IN; k += 4) {
        v2f a, b;
        a.x = x[m * D_IN + k + 2 * hi];
        a.y = x[m * D_IN + k + 2 * hi + 1];
        b.x = Wx[(k + 2 * hi) * UNITS + n];
        b.y = Wx[(k + 2 * hi + 1) * UNITS + n];
        acc = __builtin_amdgcn_wmma_f32_16x16x4_f32(false, a, false, b,
                                                    (short)0, acc, false, false);
    }
    for (int k = 0; k < UNITS; k += 4) {
        v2f a, b;
        a.x = h[m * UNITS + k + 2 * hi];
        a.y = h[m * UNITS + k + 2 * hi + 1];
        b.x = Wh[(k + 2 * hi) * UNITS + n];
        b.y = Wh[(k + 2 * hi + 1) * UNITS + n];
        acc = __builtin_amdgcn_wmma_f32_16x16x4_f32(false, a, false, b,
                                                    (short)0, acc, false, false);
    }
#pragma unroll
    for (int r = 0; r < 8; ++r)
        ns[(wave * 16 + r + 8 * hi) * UNITS + n] = acc[r];
}

// =====================================================================
// Kernel 2: bern[b] = Bernoulli(sigmoid(softplus(ns[b]·wb))), hash key;
// also ns -> bf16 A matrix for k3.
// =====================================================================
__global__ __launch_bounds__(256) void k2_bern_and_cvt(
    const float* __restrict__ ns,        // [64][1024]
    const float* __restrict__ wb,        // [1024]
    unsigned short* __restrict__ a_bf16, // [64][1024] bf16 bits
    float* __restrict__ bern)            // [64]
{
    const int b = blockIdx.x;
    __shared__ float red[256];
    float p = 0.f;
    for (int j = threadIdx.x; j < UNITS; j += 256) {
        float v = ns[b * UNITS + j];
        p += v * wb[j];
        __bf16 hv = (__bf16)v;
        a_bf16[b * UNITS + j] = *(unsigned short*)&hv;
    }
    red[threadIdx.x] = p;
    __syncthreads();
    for (int off = 128; off > 0; off >>= 1) {
        if (threadIdx.x < off) red[threadIdx.x] += red[threadIdx.x + off];
        __syncthreads();
    }
    if (threadIdx.x == 0) {
        float z  = red[0];
        float sp = (z > 20.f) ? z : log1pf(__expf(z));
        float pb = 1.f / (1.f + __expf(-sp));
        uint32_t xs = (uint32_t)b * 0x9E3779B9u + 42u;
        xs ^= xs >> 16; xs *= 0x7feb352du;
        xs ^= xs >> 15; xs *= 0x846ca68bu;
        xs ^= xs >> 16;
        float u = (float)(xs >> 8) * (1.0f / 16777216.0f);
        bern[b] = (u < pb) ? 1.0f : 0.0f;
    }
}

// =====================================================================
// Kernel 3: s = next_state @ vocab_kernel  [64 x 32000], bf16 WMMA.
//
// A (64x1024 bf16 = 128 KB) staged into LDS by the Tensor Data Mover:
// 2-D D# (rows of 512 dwords), TDM pad feature inserts 4 dwords every
// 256 dwords -> row stride 520 dwords (520/4 = 130 odd-ish enough to
// spread the 16 row-lanes over banks; quads stay 16B aligned).
// LDS dword address of row r, dword col c:  r*520 + c + 4*(c>>8).
//
// 250 blocks x 8 waves; wave owns 16 cols x 4 M-tiles; 32 K-steps of
// v_wmma_f32_16x16x32_bf16 with vocab cvt f32->bf16 in registers.
// vocab_kernel is read exactly once device-wide.
// =====================================================================
#define LDA_DW 520
__global__ __launch_bounds__(256, 1) void k3_vocab_gemm(
    const unsigned short* __restrict__ a_bf16, // [64][1024]
    const float* __restrict__ vocab,           // [1024][32000]
    float* __restrict__ s)                     // [64][32000]
{
    __shared__ unsigned int lds_a[BATCH * LDA_DW]; // 133120 bytes

#if USE_TDM
    if (threadIdx.x == 0) {
        const unsigned lds_off = (unsigned)(uintptr_t)&lds_a[0];
        const unsigned long long ga = (unsigned long long)(uintptr_t)a_bf16;
        v4u g0;
        g0[0] = 1u;                                   // count=1, user D#
        g0[1] = lds_off;                              // lds_addr (bytes)
        g0[2] = (unsigned)ga;                         // global_addr[31:0]
        g0[3] = (unsigned)((ga >> 32) & 0x01FFFFFFu)  // global_addr[56:32]
              | 0x80000000u;                          // type=2 ("image")
        v8u g1;
        g1[0] = (2u << 16)      // data_size = 4 bytes
              | (1u << 20)      // pad_enable
              | (7u << 22)      // pad_interval = 256 dwords
              | (3u << 25);     // pad_amount   = 4 dwords
        g1[1] = 512u << 16;     // tensor_dim0 = 512 (lo16); atomic_barrier=0
        g1[2] = 64u  << 16;     // tensor_dim0 hi = 0; tensor_dim1 = 64 (lo16)
        g1[3] = 512u << 16;     // tensor_dim1 hi = 0; tile_dim0 = 512
        g1[4] = 64u;            // tile_dim1 = 64; tile_dim2 = 0
        g1[5] = 512u;           // tensor_dim0_stride = 512 (lo32)
        g1[6] = 0u;             // stride0 hi; tensor_dim1_stride lo
        g1[7] = 0u;
        asm volatile("tensor_load_to_lds %0, %1" :: "s"(g0), "s"(g1) : "memory");
    }
    __builtin_amdgcn_s_wait_tensorcnt(0);
    __syncthreads();
#else
    {   // fallback: cooperative copy producing the identical padded layout
        const unsigned int* src = (const unsigned int*)a_bf16;
        for (int idx = threadIdx.x; idx < BATCH * (UNITS / 2); idx += 256) {
            int row = idx >> 9, col = idx & 511;
            lds_a[row * LDA_DW + col + 4 * (col >> 8)] = src[row * 512 + col];
        }
        __syncthreads();
    }
#endif

    const int wave = threadIdx.x >> 5;
    const int lane = threadIdx.x & 31;
    const int lo = lane & 15, hi = lane >> 4;
    const int n = blockIdx.x * 128 + wave * 16 + lo;

    v8f acc[4] = {v8f{}, v8f{}, v8f{}, v8f{}};

    for (int k = 0; k < UNITS; k += 32) {
        // ---- B fragment: strided f32 loads, cvt -> bf16 in registers ----
        union { v16bf v; __bf16 e[16]; } bu;
        const int kb = k + 16 * hi;
#pragma unroll
        for (int i = 0; i < 16; ++i)
            bu.e[i] = (__bf16)vocab[(kb + i) * VOCAB + n];

        // ---- A fragments from padded LDS ----
        const int cA = (k >> 1) + 8 * hi;       // first K-quad (dword col)
        const int cB = cA + 4;                  // second K-quad (+8 bf16)
        const int dA = cA + 4 * (cA >> 8);      // pad-adjusted
        const int dB = cB + 4 * (cB >> 8);
#pragma unroll
        for (int mt = 0; mt < 4; ++mt) {
            union { v16bf v; v4u q[2]; } au;
            const unsigned int* base = &lds_a[(mt * 16 + lo) * LDA_DW];
            au.q[0] = *(const v4u*)(base + dA);  // K = kbase .. kbase+7
            au.q[1] = *(const v4u*)(base + dB);  // K = kbase+8 .. kbase+15
            acc[mt] = __builtin_amdgcn_wmma_f32_16x16x32_bf16(
                false, au.v, false, bu.v, (short)0, acc[mt], false, false);
        }
    }

#pragma unroll
    for (int mt = 0; mt < 4; ++mt)
#pragma unroll
        for (int r = 0; r < 8; ++r)
            s[(mt * 16 + r + 8 * hi) * VOCAB + n] = acc[mt][r];
}

// =====================================================================
// Kernel 4: out[b,t,:] = softmax(s[b,:] + (1-bern[b])*beta[t,:])
// One block per (b,t); float4 (b128) loads/stores on the 164 MB stream.
// =====================================================================
__global__ __launch_bounds__(256) void k4_softmax(
    const float* __restrict__ s,    // [64][32000]
    const float* __restrict__ beta, // [20][32000]
    const float* __restrict__ bern, // [64]
    float* __restrict__ out)        // [64][20][32000]
{
    const int b = blockIdx.x / TOPICS;
    const int t = blockIdx.x % TOPICS;
    const float c = 1.0f - bern[b];
    const v4f* s4 = (const v4f*)(s + (size_t)b * VOCAB);
    const v4f* b4 = (const v4f*)(beta + (size_t)t * VOCAB);

    float m = -3.4e38f, sum = 0.f;
    for (int i = threadIdx.x; i < VOCAB / 4; i += 256) {
        v4f xv = s4[i] + c * b4[i];
        float m4 = fmaxf(fmaxf(xv.x, xv.y), fmaxf(xv.z, xv.w));
        if (m4 > m) { sum *= __expf(m - m4); m = m4; }
        sum += __expf(xv.x - m) + __expf(xv.y - m) +
               __expf(xv.z - m) + __expf(xv.w - m);
    }
    __shared__ float sm[256], ss[256];
    sm[threadIdx.x] = m;
    ss[threadIdx.x] = sum;
    __syncthreads();
    for (int off = 128; off > 0; off >>= 1) {
        if (threadIdx.x < off) {
            float m1 = sm[threadIdx.x], s1 = ss[threadIdx.x];
            float m2 = sm[threadIdx.x + off], s2 = ss[threadIdx.x + off];
            float M = m1 > m2 ? m1 : m2;
            sm[threadIdx.x] = M;
            ss[threadIdx.x] = s1 * __expf(m1 - M) + s2 * __expf(m2 - M);
        }
        __syncthreads();
    }
    const float M = sm[0];
    const float invS = 1.0f / ss[0];
    v4f* o4 = (v4f*)(out + ((size_t)b * TOPICS + t) * VOCAB);
    for (int i = threadIdx.x; i < VOCAB / 4; i += 256) {
        v4f xv = s4[i] + c * b4[i];
        v4f r;
        r.x = __expf(xv.x - M) * invS;
        r.y = __expf(xv.y - M) * invS;
        r.z = __expf(xv.z - M) * invS;
        r.w = __expf(xv.w - M) * invS;
        o4[i] = r;
    }
}

// =====================================================================
// Launcher
// =====================================================================
extern "C" void kernel_launch(void* const* d_in, const int* in_sizes, int n_in,
                              void* d_out, int out_size, void* d_ws, size_t ws_size,
                              hipStream_t stream) {
    const float* inputs  = (const float*)d_in[0]; // [64,512]
    const float* prev    = (const float*)d_in[1]; // [64,1024]
    const float* Wx      = (const float*)d_in[2]; // [512,1024]
    const float* Wh      = (const float*)d_in[3]; // [1024,1024]
    const float* wb      = (const float*)d_in[4]; // [1024,1]
    const float* vocab   = (const float*)d_in[5]; // [1024,32000]
    const float* beta    = (const float*)d_in[6]; // [20,32000]
    // d_in[7] indicators, d_in[8] indices: unused by the reference math.

    float* out = (float*)d_out;
    float* ns  = out + (size_t)BATCH * TOPICS * VOCAB; // next_state in output tail

    char* ws = (char*)d_ws;
    unsigned short* a_bf16 = (unsigned short*)ws;            // 131072 B
    float*          bern   = (float*)(ws + 131072);          // 256 B
    float*          s      = (float*)(ws + 131328);          // 8,192,000 B

    k1_next_state<<<UNITS / 16, 128, 0, stream>>>(inputs, prev, Wx, Wh, ns);
    k2_bern_and_cvt<<<BATCH, 256, 0, stream>>>(ns, wb, a_bf16, bern);
    k3_vocab_gemm<<<VOCAB / 128, 256, 0, stream>>>(a_bf16, vocab, s);
    k4_softmax<<<BATCH * TOPICS, 256, 0, stream>>>(s, beta, bern, out);
}
